// Net_81535659147677
// MI455X (gfx1250) — matrix operand order repaired and checked by reading, not compiled, not run on hardware
//
#include <hip/hip_runtime.h>
#include <hip/hip_bf16.h>

typedef unsigned short u16;
typedef __attribute__((ext_vector_type(8)))  unsigned short v8u;
typedef __attribute__((ext_vector_type(16))) unsigned short v16u;
typedef __attribute__((ext_vector_type(16))) __bf16        v16bf;
typedef __attribute__((ext_vector_type(8)))  float         v8f;
typedef __attribute__((ext_vector_type(2)))  unsigned int  v2u32;

#define T_SEQ   200
#define BATCH   4096
#define EMBD    100
#define HID     300
#define HT      19      // ceil(300/16) output h-tiles
#define KC_IH   4       // ceil(100/32) k-chunks for input proj (pad to 128)
#define KC_HH   10      // 300 padded to 320 -> 10 k-chunks
#define H1      256
#define H1T     16      // 256/16 tiles for W1 GEMM
#define NTHREADS 256    // 8 waves: 2-3 output tiles per wave -> short per-step chain
#define NWAVES   8
#define XE_STRIDE 136   // halves; 272B rows: 16B aligned, 68 dwords -> conflict-free
#define H_STRIDE  328   // halves; 656B rows: 16B aligned, 164 dwords -> conflict-free
#define H1_STRIDE 260   // floats; conflict-free for strided f32 reads

// Hardware tanh (V_TANH_F32) if the builtin exists on this toolchain.
#if defined(__has_builtin)
#if __has_builtin(__builtin_amdgcn_tanhf)
#define TANHF(x) __builtin_amdgcn_tanhf(x)
#endif
#endif
#ifndef TANHF
#define TANHF(x) tanhf(x)
#endif

// f32 -> bf16 round-to-nearest-even
__device__ __forceinline__ u16 f2bf(float f) {
  unsigned u = __builtin_bit_cast(unsigned, f);
  u += 0x7FFFu + ((u >> 16) & 1u);
  return (u16)(u >> 16);
}

// Load one 16x32 bf16 A-matrix fragment (16 halves) for this lane from LDS.
// p points at row base + chunk*32 + (lane>=16 ? 8 : 0). Per ISA layout the lane
// holds K = [k0..k0+7] and [k0+16..k0+23]: two 16B ds_load_b128.
__device__ __forceinline__ v16bf load_a(const u16* p) {
  v8u lo = *(const v8u*)(p);
  v8u hi = *(const v8u*)(p + 16);
  v16u r = __builtin_shufflevector(lo, hi, 0,1,2,3,4,5,6,7,8,9,10,11,12,13,14,15);
  return __builtin_bit_cast(v16bf, r);
}

// Pack W (row-major [N,K], f32) into the WMMA B-operand layout, bf16:
// B[k][n] = W[n][k]; record per (tile, chunk, lane) = 16 contiguous halves.
// Lane L: n = L&15, kbase = (L<16 ? 0 : 8); half j -> K = chunk*32+kbase+j+(j>=8 ? 8 : 0).
__global__ void pack_b_kernel(const float* __restrict__ W, int N, int K,
                              int nchunks, u16* __restrict__ out) {
  int tile  = blockIdx.x / nchunks;
  int lane  = threadIdx.x >> 4;
  int j     = threadIdx.x & 15;
  int chunk = blockIdx.x - tile * nchunks;
  int n     = lane & 15;
  int kbase = (lane < 16) ? 0 : 8;
  int row   = tile * 16 + n;
  int k     = chunk * 32 + kbase + j + ((j >= 8) ? 8 : 0);
  float v = (row < N && k < K) ? W[(size_t)row * K + k] : 0.f;
  out[((size_t)blockIdx.x * 32 + lane) * 16 + j] = f2bf(v);
}

__global__ void bias_kernel(const float* __restrict__ b_ih,
                            const float* __restrict__ b_hh,
                            float* __restrict__ bsum) {
  int i = threadIdx.x;
  if (i < 304) bsum[i] = (i < HID) ? (b_ih[i] + b_hh[i]) : 0.f;
}

// Persistent RNN kernel: each block owns 16 batch rows for the whole scan.
__global__ __launch_bounds__(NTHREADS) void rnn_kernel(
    const int* __restrict__ x, const float* __restrict__ emb,
    const u16* __restrict__ Bih, const u16* __restrict__ Bhh,
    const u16* __restrict__ BW1, const float* __restrict__ bsum,
    const float* __restrict__ b1, const float* __restrict__ W2,
    const float* __restrict__ b2, float* __restrict__ out) {
  __shared__ u16   s_xe[16 * XE_STRIDE];     // embedded inputs (bf16), e padded to 128
  __shared__ u16   s_h[2][16 * H_STRIDE];    // ping-pong hidden state (bf16), h padded to 320
  __shared__ float s_h1[16 * H1_STRIDE];     // relu(h@W1^T+b1) for the head
  __shared__ int   s_tok[16];

  const int tid   = threadIdx.x;
  const int wave  = tid >> 5;
  const int lane  = tid & 31;
  const int lrow  = lane & 15;   // A-row (batch) / D-col (h) index within tile
  const int lhi   = lane >> 4;   // half-wave select
  const int khalf = lhi ? 8 : 0; // A fragment K sub-offset

  for (int i = tid; i < 16 * H_STRIDE; i += NTHREADS) { s_h[0][i] = 0; s_h[1][i] = 0; }
  for (int i = tid; i < 16 * XE_STRIDE; i += NTHREADS) s_xe[i] = 0;

  const v16u* vBih = (const v16u*)Bih;
  const v16u* vBhh = (const v16u*)Bhh;
  const v16u* vBW1 = (const v16u*)BW1;

  for (int t = 0; t < T_SEQ; ++t) {
    const int cur = t & 1, nxt = cur ^ 1;
    if (tid < 16) s_tok[tid] = x[t * BATCH + blockIdx.x * 16 + tid];
    __syncthreads();  // also orders prev-step compute before xe overwrite
    // Vectorized gather: 16 rows x 25 float4 groups (e = 0..99); padded tail
    // (e >= 100) stays zero from init. Row stride 400B and group stride 16B
    // keep float4 loads aligned; LDS store is one 8B packed-bf16 write.
    for (int i = tid; i < 16 * 25; i += NTHREADS) {
      int r = i / 25, e4 = i - r * 25;
      float4 v = ((const float4*)(emb + (size_t)s_tok[r] * EMBD))[e4];
      v2u32 p;
      p.x = (unsigned)f2bf(v.x) | ((unsigned)f2bf(v.y) << 16);
      p.y = (unsigned)f2bf(v.z) | ((unsigned)f2bf(v.w) << 16);
      *(v2u32*)&s_xe[r * XE_STRIDE + e4 * 4] = p;
    }
    __syncthreads();

    for (int tile = wave; tile < HT; tile += NWAVES) {
      const float bias = bsum[tile * 16 + lrow];  // bsum zero-padded to 304
      v8f c;
      #pragma unroll
      for (int r = 0; r < 8; ++r) c[r] = bias;
      #pragma unroll
      for (int kc = 0; kc < KC_IH; ++kc) {        // xe @ W_ih^T
        v16bf a = load_a(&s_xe[lrow * XE_STRIDE + kc * 32 + khalf]);
        v16bf b = __builtin_bit_cast(v16bf, vBih[(size_t)(tile * KC_IH + kc) * 32 + lane]);
        c = __builtin_amdgcn_wmma_f32_16x16x32_bf16(false, a, false, b, (short)0, c, false, false);
      }
      #pragma unroll
      for (int kc = 0; kc < KC_HH; ++kc) {        // h @ W_hh^T
        v16bf a = load_a(&s_h[cur][lrow * H_STRIDE + kc * 32 + khalf]);
        v16bf b = __builtin_bit_cast(v16bf, vBhh[(size_t)(tile * KC_HH + kc) * 32 + lane]);
        c = __builtin_amdgcn_wmma_f32_16x16x32_bf16(false, a, false, b, (short)0, c, false, false);
      }
      #pragma unroll
      for (int r = 0; r < 8; ++r) {               // hardware tanh + store next state
        int m = r + lhi * 8;
        s_h[nxt][m * H_STRIDE + tile * 16 + lrow] = f2bf(TANHF(c[r]));
      }
    }
    __syncthreads();
  }

  // ---- fused MLP head ----
  const int fin = T_SEQ & 1;  // step t writes buffer (t+1)&1 -> final in T&1
  for (int tile = wave; tile < H1T; tile += NWAVES) {
    const float bias = b1[tile * 16 + lrow];
    v8f c;
    #pragma unroll
    for (int r = 0; r < 8; ++r) c[r] = bias;
    #pragma unroll
    for (int kc = 0; kc < KC_HH; ++kc) {
      v16bf a = load_a(&s_h[fin][lrow * H_STRIDE + kc * 32 + khalf]);
      v16bf b = __builtin_bit_cast(v16bf, vBW1[(size_t)(tile * KC_HH + kc) * 32 + lane]);
      c = __builtin_amdgcn_wmma_f32_16x16x32_bf16(false, a, false, b, (short)0, c, false, false);
    }
    #pragma unroll
    for (int r = 0; r < 8; ++r) {
      int m = r + lhi * 8;
      s_h1[m * H1_STRIDE + tile * 16 + lrow] = fmaxf(c[r], 0.f);
    }
  }
  __syncthreads();
  if (tid < 32) {  // final [16,256] x [256,2] -> one output element per lane
    int b = tid & 15, j = tid >> 4;
    float acc = b2[j];
    for (int k = 0; k < H1; ++k) acc += s_h1[b * H1_STRIDE + k] * W2[j * H1 + k];
    out[((size_t)blockIdx.x * 16 + b) * 2 + j] = acc;
  }
}

extern "C" void kernel_launch(void* const* d_in, const int* in_sizes, int n_in,
                              void* d_out, int out_size, void* d_ws, size_t ws_size,
                              hipStream_t stream) {
  (void)in_sizes; (void)n_in; (void)out_size; (void)ws_size;
  const int*   x    = (const int*)d_in[0];
  const float* emb  = (const float*)d_in[1];
  const float* W_ih = (const float*)d_in[2];
  const float* b_ih = (const float*)d_in[3];
  const float* W_hh = (const float*)d_in[4];
  const float* b_hh = (const float*)d_in[5];
  const float* W1   = (const float*)d_in[6];
  const float* b1   = (const float*)d_in[7];
  const float* W2   = (const float*)d_in[8];
  const float* b2   = (const float*)d_in[9];
  float* out = (float*)d_out;

  char* ws = (char*)d_ws;
  float* bsum = (float*)(ws);                            //   1,216 B (pad to 4096)
  u16*   Bih  = (u16*)(ws + 4096);                       //  77,824 B
  u16*   Bhh  = (u16*)(ws + 4096 + 77824);               // 194,560 B
  u16*   BW1  = (u16*)(ws + 4096 + 77824 + 194560);      // 163,840 B  (total < 1 MB)

  bias_kernel<<<1, 512, 0, stream>>>(b_ih, b_hh, bsum);
  pack_b_kernel<<<HT * KC_IH, 512, 0, stream>>>(W_ih, HID, EMBD, KC_IH, Bih);
  pack_b_kernel<<<HT * KC_HH, 512, 0, stream>>>(W_hh, HID, HID, KC_HH, Bhh);
  pack_b_kernel<<<H1T * KC_HH, 512, 0, stream>>>(W1, H1, HID, KC_HH, BW1);
  rnn_kernel<<<BATCH / 16, NTHREADS, 0, stream>>>(x, emb, Bih, Bhh, BW1, bsum, b1, W2, b2, out);
}
